// LazyCrossAttentionGQA_4612794876258
// MI455X (gfx1250) — compile-verified, matrix operand order checked by reading, not run on hardware
//
#include <hip/hip_runtime.h>
#include <math.h>

// ---------------------------------------------------------------------------
// Types for CDNA5 WMMA (wave32): D(16x16 f32) = A(16x32 bf16) x B(32x16 bf16) + C
// ---------------------------------------------------------------------------
typedef __attribute__((ext_vector_type(16))) __bf16 v16bf;
typedef __attribute__((ext_vector_type(8)))  __bf16 v8bf;
typedef __attribute__((ext_vector_type(8)))  float  v8f;

#define TQ      2048
#define TKC     2048
#define DMODEL  2048
#define DHEAD   128
#define GKVN    4
#define HQ      16
#define REPS    1e-6f

// ---------------------------------------------------------------------------
// A/B fragment loader. Per CDNA5 ISA 16-bit A/B layout (wave32):
//   lane L (row/col = L%16) holds K = kb+0..7 (VGPR0-3) and K = kb+16..23
//   (VGPR4-7), kb = 8*(L/16). With base = rowptr + k0 + kb the fragment is two
//   contiguous 16-byte chunks at base and base+16 elements. Works for global
//   (global_load_b128) and LDS (ds_load_b128) pointers.
// ---------------------------------------------------------------------------
static __device__ __forceinline__ v16bf load_frag(const __bf16* base) {
  v8bf lo = *(const v8bf*)(base);
  v8bf hi = *(const v8bf*)(base + 16);
  v16bf f;
#pragma unroll
  for (int i = 0; i < 8; ++i) { f[i] = lo[i]; f[8 + i] = hi[i]; }
  return f;
}

// ---------------------------------------------------------------------------
// Kernel 1: RMSNorm over last dim (D=2048) + cast to bf16. One block per row.
// ---------------------------------------------------------------------------
__global__ void k_rmsnorm_cast(const float* __restrict__ x,
                               const float* __restrict__ nw,
                               __bf16* __restrict__ xn) {
  __shared__ float red[8];
  const int row = blockIdx.x;
  const float* xr = x + (size_t)row * DMODEL;
  float ss = 0.f;
#pragma unroll
  for (int i = 0; i < 8; ++i) {
    float v = xr[threadIdx.x + i * 256];
    ss += v * v;
  }
#pragma unroll
  for (int off = 16; off > 0; off >>= 1) ss += __shfl_down(ss, off, 32);
  const int lane = threadIdx.x & 31, wave = threadIdx.x >> 5;
  if (lane == 0) red[wave] = ss;
  __syncthreads();
  if (threadIdx.x == 0) {
    float t = 0.f;
#pragma unroll
    for (int i = 0; i < 8; ++i) t += red[i];
    red[0] = rsqrtf(t / (float)DMODEL + REPS);
  }
  __syncthreads();
  const float rs = red[0];
#pragma unroll
  for (int i = 0; i < 8; ++i) {
    int c = threadIdx.x + i * 256;
    xn[(size_t)row * DMODEL + c] = (__bf16)(xr[c] * rs * nw[c]);
  }
}

// ---------------------------------------------------------------------------
// Kernel 2: elementwise fp32 -> bf16 cast (w_q / w_out; 8MB each => L2-resident)
// ---------------------------------------------------------------------------
__global__ void k_cast_bf16(const float* __restrict__ src,
                            __bf16* __restrict__ dst, int n) {
  int i = blockIdx.x * 256 + threadIdx.x;
  if (i < n) dst[i] = (__bf16)src[i];
}

// ---------------------------------------------------------------------------
// Kernel 3: repack K -> (B,G,Tk,d) bf16 and V -> V^T (B,G,d,Tk) bf16 so that
// both QK^T B-fragments and PV B-fragments are contiguous along K.
// ---------------------------------------------------------------------------
__global__ void k_repack_kv(const float* __restrict__ k_ctx,
                            const float* __restrict__ v_ctx,
                            __bf16* __restrict__ kb,
                            __bf16* __restrict__ vt, int Bb) {
  size_t i = (size_t)blockIdx.x * 256 + threadIdx.x;
  size_t total = (size_t)Bb * TKC * GKVN * DHEAD;
  if (i >= total) return;
  int j = (int)(i % DHEAD);
  size_t t2 = i / DHEAD;
  int g = (int)(t2 % GKVN);
  size_t t3 = t2 / GKVN;
  int t = (int)(t3 % TKC);
  int b = (int)(t3 / TKC);
  kb[(((size_t)(b * GKVN + g)) * TKC + t) * DHEAD + j] = (__bf16)k_ctx[i];
  vt[(((size_t)(b * GKVN + g)) * DHEAD + j) * TKC + t] = (__bf16)v_ctx[i];
}

// ---------------------------------------------------------------------------
// Kernel 4/6: C[m,n] = sum_k A[m,k] * W[n,k]  (torch Linear: y = x @ W^T)
// Block = 256 threads = 8 waves; block tile 128x128; wave tile 32x64
// (2 M-frags x 4 N-frags), K-step 32 -> 8 wmma per wave per step.
// Register double-buffering: fragments for k0+32 are in flight while the
// 8 WMMAs for k0 execute; L2 prefetch one further step ahead.
// __launch_bounds__(256,1): min 1 wave/SIMD -> full VGPR budget, no spills.
// MODE 0: q-projection epilogue: scale, cast bf16, scatter to (B,H,Tq,d).
// MODE 1: fp32 output row-major (final result).
// ---------------------------------------------------------------------------
template <int MODE>
__global__ void __launch_bounds__(256, 1)
k_gemm_bf16_nt(const __bf16* __restrict__ A,
               const __bf16* __restrict__ W,
               void* __restrict__ out,
               int M, int N, int K, float scale) {
  const int lane = threadIdx.x & 31;
  const int wave = threadIdx.x >> 5;
  const int wm = wave >> 1;          // 0..3
  const int wn = wave & 1;           // 0..1
  const int lg = lane >> 4;          // half-wave
  const int ln = lane & 15;
  const int row0 = blockIdx.x * 128 + wm * 32;
  const int col0 = blockIdx.y * 128 + wn * 64;
  const int kb8 = lg * 8;

  v8f acc[2][4] = {};

  const __bf16* arow[2];
  arow[0] = A + (size_t)(row0 + ln) * K + kb8;
  arow[1] = A + (size_t)(row0 + 16 + ln) * K + kb8;
  const __bf16* wrow[4];
#pragma unroll
  for (int nt = 0; nt < 4; ++nt)
    wrow[nt] = W + (size_t)(col0 + nt * 16 + ln) * K + kb8;

  v16bf af[2][2], bfr[2][4];
  af[0][0] = load_frag(arow[0]);
  af[0][1] = load_frag(arow[1]);
#pragma unroll
  for (int nt = 0; nt < 4; ++nt) bfr[0][nt] = load_frag(wrow[nt]);

  for (int k0 = 0; k0 < K; k0 += 32) {
    const int cur = (k0 >> 5) & 1;
    const int nxt = cur ^ 1;
    const int kn = k0 + 32;
    if (kn < K) {
      af[nxt][0] = load_frag(arow[0] + kn);
      af[nxt][1] = load_frag(arow[1] + kn);
#pragma unroll
      for (int nt = 0; nt < 4; ++nt) bfr[nxt][nt] = load_frag(wrow[nt] + kn);
      if (kn + 32 < K) {  // pull following step into cache (global_prefetch_b8)
        __builtin_prefetch(arow[0] + kn + 32, 0, 1);
        __builtin_prefetch(wrow[0] + kn + 32, 0, 1);
      }
    }
#pragma unroll
    for (int mt = 0; mt < 2; ++mt)
#pragma unroll
      for (int nt = 0; nt < 4; ++nt)
        acc[mt][nt] = __builtin_amdgcn_wmma_f32_16x16x32_bf16(
            false, af[cur][mt], false, bfr[cur][nt], (short)0, acc[mt][nt],
            false, false);
  }

  // C layout: lane holds n = ln, m = r + 8*lg within each 16x16 tile.
#pragma unroll
  for (int mt = 0; mt < 2; ++mt) {
#pragma unroll
    for (int nt = 0; nt < 4; ++nt) {
#pragma unroll
      for (int rr = 0; rr < 8; ++rr) {
        int m = row0 + mt * 16 + rr + 8 * lg;
        int n = col0 + nt * 16 + ln;
        float v = acc[mt][nt][rr];
        if (MODE == 0) {
          int b = m / TQ, t = m % TQ;
          int h = n / DHEAD, j = n % DHEAD;
          ((__bf16*)out)[(((size_t)(b * HQ + h)) * TQ + t) * DHEAD + j] =
              (__bf16)(v * scale);
        } else {
          ((float*)out)[(size_t)m * N + n] = v;
        }
      }
    }
  }
}

// ---------------------------------------------------------------------------
// Kernel 5: flash attention (GQA). grid = (B*H, Tq/128), block = 256 (8 waves).
// All 8 waves share one K/V stream -> K and V^T chunks are staged in LDS
// cooperatively (8x VMEM reduction) with double buffering: next chunk's global
// loads are issued at the top of the chunk, parked in registers across only
// the QK^T WMMA phase, then committed to the alternate LDS buffer (short live
// range -> no spills). Fragments feed the WMMAs from LDS via ds_load_b128.
// Softmax scale is pre-folded into q.
// ---------------------------------------------------------------------------
__global__ void __launch_bounds__(256, 1)
k_flash_attn(const __bf16* __restrict__ q,
             const __bf16* __restrict__ kb,
             const __bf16* __restrict__ vt,
             __bf16* __restrict__ ctx) {
  const int bh = blockIdx.x;
  const int b = bh / HQ, h = bh % HQ;
  const int g = h / (HQ / GKVN);
  const int tid  = threadIdx.x;
  const int lane = tid & 31;
  const int wave = tid >> 5;
  const int lg = lane >> 4;
  const int ln = lane & 15;
  const int kb8 = lg * 8;
  const int qbase = blockIdx.y * 128 + wave * 16;

  __shared__ __align__(16) __bf16 Kt[2][64][DHEAD];   // keys chunk, 2x16KB
  __shared__ __align__(16) __bf16 Vt[2][DHEAD][64];   // V^T chunk,  2x16KB
  __shared__ float  S[8][16][64];                     // 32KB
  __shared__ __align__(16) __bf16 P[8][16][64];       // 16KB
  __shared__ float rowm[8][16], rowl[8][16], rowf[8][16];

  const __bf16* qptr = q  + ((size_t)(b * HQ + h))   * TQ  * DHEAD;
  const __bf16* kptr = kb + ((size_t)(b * GKVN + g)) * TKC * DHEAD;
  const __bf16* vtp  = vt + ((size_t)(b * GKVN + g)) * DHEAD * TKC;

  // Q fragments: 4 K-steps covering d=128, loaded once.
  v16bf qa[4];
#pragma unroll
  for (int ks = 0; ks < 4; ++ks)
    qa[ks] = load_frag(qptr + (size_t)(qbase + ln) * DHEAD + ks * 32 + kb8);

  v8f o[8] = {};  // ctx accumulator: 8 N-tiles over d=128

  if (lane < 16) { rowm[wave][lane] = -1e30f; rowl[wave][lane] = 0.f; }

  // stage chunk 0 into buffer 0 (64x128 K rows, 128x64 V^T rows; 4 b128 each)
#pragma unroll
  for (int i = 0; i < 4; ++i) {
    int u = tid + i * 256;
    int kr = u >> 4, kc = (u & 15) * 8;
    *(v8bf*)&Kt[0][kr][kc] = *(const v8bf*)(kptr + (size_t)kr * DHEAD + kc);
    int vr = u >> 3, vc = (u & 7) * 8;
    *(v8bf*)&Vt[0][vr][vc] = *(const v8bf*)(vtp + (size_t)vr * TKC + vc);
  }

  int buf = 0;
  for (int tk0 = 0; tk0 < TKC; tk0 += 64) {
    __syncthreads();  // staged chunk visible; previous chunk's readers done

    // issue next chunk's global loads now; park in registers (overlaps QK^T)
    v8bf kreg[4], vreg[4];
    const bool have_next = (tk0 + 64) < TKC;
    if (have_next) {
#pragma unroll
      for (int i = 0; i < 4; ++i) {
        int u = tid + i * 256;
        int kr = u >> 4, kc = (u & 15) * 8;
        kreg[i] = *(const v8bf*)(kptr + (size_t)(tk0 + 64 + kr) * DHEAD + kc);
        int vr = u >> 3, vc = (u & 7) * 8;
        vreg[i] = *(const v8bf*)(vtp + (size_t)vr * TKC + tk0 + 64 + vc);
      }
      if (tk0 + 128 < TKC) {  // L2 prefetch two chunks ahead
        __builtin_prefetch(
            kptr + (size_t)(tk0 + 128 + (tid >> 2)) * DHEAD + (tid & 3) * 32, 0, 1);
        __builtin_prefetch(
            vtp + (size_t)(tid >> 1) * TKC + tk0 + 128 + (tid & 1) * 32, 0, 1);
      }
    }

    // ---- S = Q K^T for a 16x64 tile (16 WMMAs, B-frags from LDS) ----
    v8f sacc[4] = {};
#pragma unroll
    for (int nt = 0; nt < 4; ++nt)
#pragma unroll
      for (int ks = 0; ks < 4; ++ks) {
        v16bf bfrag = load_frag(&Kt[buf][nt * 16 + ln][ks * 32 + kb8]);
        sacc[nt] = __builtin_amdgcn_wmma_f32_16x16x32_bf16(
            false, qa[ks], false, bfrag, (short)0, sacc[nt], false, false);
      }

    // ---- commit parked next chunk to the alternate LDS buffer now:
    //      buf^1 is not read by any wave until after the next barrier, and
    //      retiring kreg/vreg here keeps peak VGPR pressure low. ----
    if (have_next) {
#pragma unroll
      for (int i = 0; i < 4; ++i) {
        int u = tid + i * 256;
        int kr = u >> 4, kc = (u & 15) * 8;
        *(v8bf*)&Kt[buf ^ 1][kr][kc] = kreg[i];
        int vr = u >> 3, vc = (u & 7) * 8;
        *(v8bf*)&Vt[buf ^ 1][vr][vc] = vreg[i];
      }
    }

#pragma unroll
    for (int nt = 0; nt < 4; ++nt)
#pragma unroll
      for (int rr = 0; rr < 8; ++rr)
        S[wave][rr + 8 * lg][nt * 16 + ln] = sacc[nt][rr];
    __syncthreads();

    // ---- online softmax: lanes 0..15 own one row each ----
    if (lane < 16) {
      float mold = rowm[wave][lane];
      float cm = mold;
#pragma unroll 8
      for (int jj = 0; jj < 64; ++jj) cm = fmaxf(cm, S[wave][lane][jj]);
      float fac = __expf(mold - cm);
      float s = 0.f;
#pragma unroll 8
      for (int jj = 0; jj < 64; ++jj) {
        float p = __expf(S[wave][lane][jj] - cm);
        s += p;
        P[wave][lane][jj] = (__bf16)p;
      }
      rowm[wave][lane] = cm;
      rowl[wave][lane] = rowl[wave][lane] * fac + s;
      rowf[wave][lane] = fac;
    }
    __syncthreads();

    // ---- rescale running accumulators by per-row factor ----
#pragma unroll
    for (int rr = 0; rr < 8; ++rr) {
      float fac = rowf[wave][rr + 8 * lg];
#pragma unroll
      for (int nt = 0; nt < 8; ++nt) o[nt][rr] *= fac;
    }

    // ---- o += P @ V  (P: A-frags from LDS; V^T: B-frags from LDS) ----
    v16bf pa[2];
#pragma unroll
    for (int ks = 0; ks < 2; ++ks)
      pa[ks] = load_frag(&P[wave][ln][ks * 32 + kb8]);
#pragma unroll
    for (int nt = 0; nt < 8; ++nt)
#pragma unroll
      for (int ks = 0; ks < 2; ++ks) {
        v16bf bfrag = load_frag(&Vt[buf][nt * 16 + ln][ks * 32 + kb8]);
        o[nt] = __builtin_amdgcn_wmma_f32_16x16x32_bf16(
            false, pa[ks], false, bfrag, (short)0, o[nt], false, false);
      }

    buf ^= 1;
  }

  // ---- normalize + store ctx in (B,Tq,D) bf16 for the out-projection ----
#pragma unroll
  for (int rr = 0; rr < 8; ++rr) {
    int mloc = rr + 8 * lg;
    float inv = 1.f / rowl[wave][mloc];
    int t = qbase + mloc;
    size_t base = ((size_t)(b * TQ + t)) * DMODEL + h * DHEAD;
#pragma unroll
    for (int nt = 0; nt < 8; ++nt)
      ctx[base + nt * 16 + ln] = (__bf16)(o[nt][rr] * inv);
  }
}

// ---------------------------------------------------------------------------
// Host-side orchestration (all launches on `stream`, scratch carved from d_ws)
// ---------------------------------------------------------------------------
extern "C" void kernel_launch(void* const* d_in, const int* in_sizes, int n_in,
                              void* d_out, int out_size, void* d_ws, size_t ws_size,
                              hipStream_t stream) {
  const float* x_q    = (const float*)d_in[0];
  const float* k_ctx  = (const float*)d_in[1];
  const float* v_ctx  = (const float*)d_in[2];
  const float* w_q    = (const float*)d_in[3];
  const float* w_out  = (const float*)d_in[4];
  const float* norm_w = (const float*)d_in[5];

  const int Bb = in_sizes[0] / (TQ * DMODEL);  // batch (2)
  const int M  = Bb * TQ;

  // workspace layout (bf16 buffers)
  char* ws = (char*)d_ws;
  const size_t szXN = (size_t)Bb * TQ * DMODEL * 2;
  const size_t szW  = (size_t)DMODEL * DMODEL * 2;
  const size_t szKV = (size_t)Bb * GKVN * TKC * DHEAD * 2;
  __bf16* xn   = (__bf16*)(ws);
  __bf16* wqb  = (__bf16*)(ws + szXN);
  __bf16* wob  = (__bf16*)(ws + szXN + szW);
  __bf16* qb   = (__bf16*)(ws + szXN + 2 * szW);
  __bf16* kbb  = (__bf16*)(ws + 2 * szXN + 2 * szW);
  __bf16* vtb  = (__bf16*)(ws + 2 * szXN + 2 * szW + szKV);
  __bf16* ctxb = (__bf16*)(ws + 2 * szXN + 2 * szW + 2 * szKV);

  // 1) RMSNorm + bf16 cast
  k_rmsnorm_cast<<<M, 256, 0, stream>>>(x_q, norm_w, xn);

  // 2) weight casts (L2-resident afterwards)
  const int nW = DMODEL * DMODEL;
  k_cast_bf16<<<(nW + 255) / 256, 256, 0, stream>>>(w_q, wqb, nW);
  k_cast_bf16<<<(nW + 255) / 256, 256, 0, stream>>>(w_out, wob, nW);

  // 3) K / V^T repack
  const size_t nKV = (size_t)Bb * TKC * GKVN * DHEAD;
  k_repack_kv<<<(int)((nKV + 255) / 256), 256, 0, stream>>>(k_ctx, v_ctx, kbb, vtb, Bb);

  // 4) q = xn @ w_q^T, scaled by 1/sqrt(d), scattered to (B,H,Tq,d)
  const float scale = 0.08838834764831845f;  // 1/sqrt(128)
  dim3 ggrid(M / 128, DMODEL / 128);
  k_gemm_bf16_nt<0><<<ggrid, 256, 0, stream>>>(xn, wqb, (void*)qb, M, DMODEL, DMODEL, scale);

  // 5) flash attention -> ctx (B,Tq,D) bf16
  dim3 agrid(Bb * HQ, TQ / 128);
  k_flash_attn<<<agrid, 256, 0, stream>>>(qb, kbb, vtb, ctxb);

  // 6) out = ctx @ w_out^T -> fp32 d_out
  k_gemm_bf16_nt<1><<<ggrid, 256, 0, stream>>>(ctxb, wob, d_out, M, DMODEL, DMODEL, 1.0f);
}
